// TimerBackbone_1924145349256
// MI455X (gfx1250) — compile-verified, matrix operand order checked by reading, not run on hardware
//
#include <hip/hip_runtime.h>

// ---------------------------------------------------------------------------
// MI455X (gfx1250) Timer-backbone forward. bf16 WMMA everywhere; fp32 acc.
// GEMMs: cooperative 64x512 block tile, A-panel double-buffered in LDS via
// async-to-LDS DMA (ASYNCcnt), B streamed from L2 with prefetch hints.
// ---------------------------------------------------------------------------

typedef __bf16 bf16;
typedef __attribute__((ext_vector_type(16))) __bf16 v16bf;
typedef __attribute__((ext_vector_type(8)))  __bf16 v8bf;
typedef __attribute__((ext_vector_type(4)))  __bf16 v4bf;
typedef __attribute__((ext_vector_type(8)))  float  v8f;
typedef int v4i __attribute__((vector_size(16)));   // pointee type of async builtin

#define D_MODEL 1024
#define N_PATCH 1024
#define ROWS    4096   // B*V*N = 4*1*1024
#define DFF_    2048
#define NHEAD   8
#define DH_     128
#define PLEN    96
#define ELAYERS 8

enum { MODE_F32 = 0, MODE_PE = 1, MODE_QK = 2, MODE_VT = 3, MODE_RELU = 4, MODE_RESID = 5 };

// --------------------------- device helpers --------------------------------

__device__ __forceinline__ v8f vzero8() {
  v8f z;
#pragma unroll
  for (int i = 0; i < 8; ++i) z[i] = 0.f;
  return z;
}

// Per-lane 16-bit operand fragment (A 16x32 or B 32x16): two contiguous
// 16-byte chunks at +0 and +32 bytes (k = klo..klo+7 and klo+16..klo+23).
__device__ __forceinline__ v16bf ldfrag(const bf16* p) {
  v8bf lo = *(const v8bf*)(p);
  v8bf hv = *(const v8bf*)(p + 16);
  v16bf f;
#pragma unroll
  for (int i = 0; i < 8; ++i) { f[i] = lo[i]; f[8 + i] = hv[i]; }
  return f;
}

__device__ __forceinline__ v8f wmma_bf(v16bf a, v16bf b, v8f c) {
  return __builtin_amdgcn_wmma_f32_16x16x32_bf16(false, a, false, b, (short)0, c,
                                                 false, false);
}

// Async 16B global->LDS copy (one b128 chunk per lane), tracked by ASYNCcnt.
__device__ __forceinline__ void async_copy16(const bf16* gsrc, bf16* ldst) {
#if __has_builtin(__builtin_amdgcn_global_load_async_to_lds_b128)
  __builtin_amdgcn_global_load_async_to_lds_b128(
      (__attribute__((address_space(1))) v4i*)gsrc,
      (__attribute__((address_space(3))) v4i*)ldst, 0, 0);
#else
  unsigned loff = (unsigned)(size_t)(__attribute__((address_space(3))) bf16*)ldst;
  asm volatile("global_load_async_to_lds_b128 %0, %1, off"
               :: "v"(loff), "v"(gsrc) : "memory");
#endif
}

__device__ __forceinline__ void wait_async0() {
#if __has_builtin(__builtin_amdgcn_s_wait_asynccnt)
  __builtin_amdgcn_s_wait_asynccnt(0);
#else
  asm volatile("s_wait_asynccnt 0x0" ::: "memory");
#endif
}

// --------------------------- small kernels ---------------------------------

__global__ __launch_bounds__(256) void cvt_f32_bf16(const float* __restrict__ in,
                                                    bf16* __restrict__ out, int n4) {
  int i = blockIdx.x * 256 + threadIdx.x;
  if (i < n4) {
    float4 v = ((const float4*)in)[i];
    v4bf o;
    o[0] = (bf16)v.x; o[1] = (bf16)v.y; o[2] = (bf16)v.z; o[3] = (bf16)v.w;
    ((v4bf*)out)[i] = o;
  }
}

__global__ __launch_bounds__(256) void pe_kernel(float* __restrict__ pe) {
  int idx = blockIdx.x * 256 + threadIdx.x;      // N_PATCH * D_MODEL threads
  int n = idx >> 10, d = idx & 1023;
  float div = __expf((float)(d & ~1) * (-9.210340371976184f / 1024.f));
  float ang = (float)n * div;
  pe[idx] = (d & 1) ? __cosf(ang) : __sinf(ang);
}

// LayerNorm over D=1024; writes fp32 + bf16 copies. One block per row.
__global__ __launch_bounds__(256) void ln_kernel(const float* __restrict__ x,
                                                 const float* __restrict__ g,
                                                 const float* __restrict__ bb,
                                                 float* __restrict__ y,
                                                 bf16* __restrict__ ybf) {
  __shared__ float rs1[8], rs2[8], stats[2];
  const size_t row = blockIdx.x;
  const float4 v = ((const float4*)(x + row * D_MODEL))[threadIdx.x];
  float s1 = v.x + v.y + v.z + v.w;
  float s2 = v.x * v.x + v.y * v.y + v.z * v.z + v.w * v.w;
#pragma unroll
  for (int d = 1; d < 32; d <<= 1) {
    s1 += __shfl_xor(s1, d, 32);
    s2 += __shfl_xor(s2, d, 32);
  }
  const int wave = threadIdx.x >> 5, lane = threadIdx.x & 31;
  if (lane == 0) { rs1[wave] = s1; rs2[wave] = s2; }
  __syncthreads();
  if (threadIdx.x == 0) {
    float t1 = 0.f, t2 = 0.f;
#pragma unroll
    for (int i = 0; i < 8; ++i) { t1 += rs1[i]; t2 += rs2[i]; }
    float mean = t1 * (1.f / 1024.f);
    float var  = t2 * (1.f / 1024.f) - mean * mean;
    stats[0] = mean;
    stats[1] = rsqrtf(var + 1e-5f);
  }
  __syncthreads();
  const float mean = stats[0], rstd = stats[1];
  const int c0 = threadIdx.x * 4;
  float4 o;
  o.x = (v.x - mean) * rstd * g[c0 + 0] + bb[c0 + 0];
  o.y = (v.y - mean) * rstd * g[c0 + 1] + bb[c0 + 1];
  o.z = (v.z - mean) * rstd * g[c0 + 2] + bb[c0 + 2];
  o.w = (v.w - mean) * rstd * g[c0 + 3] + bb[c0 + 3];
  ((float4*)(y + row * D_MODEL))[threadIdx.x] = o;
  v4bf ob;
  ob[0] = (bf16)o.x; ob[1] = (bf16)o.y; ob[2] = (bf16)o.z; ob[3] = (bf16)o.w;
  ((v4bf*)(ybf + row * D_MODEL))[threadIdx.x] = ob;
}

// --------------------------- epilogue (used by gemm_lds_kernel) ---------------

template <int MODE>
__device__ __forceinline__ void gemm_epilogue(
    v8f acc[4][4], int m0, int n0, int lr, int hi8,
    const float* __restrict__ bias, float* __restrict__ outF,
    bf16* __restrict__ outB, const float* __restrict__ resid,
    const float* __restrict__ pe, int Nout, float alpha) {
#pragma unroll
  for (int i = 0; i < 4; ++i) {
#pragma unroll
    for (int j = 0; j < 4; ++j) {
#pragma unroll
      for (int r = 0; r < 8; ++r) {
        const int m = m0 + i * 16 + hi8 + r;
        const int n = n0 + j * 16 + lr;
        float v = acc[i][j][r];
        if (bias) v += bias[n];
        v *= alpha;
        if (MODE == MODE_F32) {
          outF[(size_t)m * Nout + n] = v;
        } else if (MODE == MODE_PE) {
          v += pe[(size_t)(m & (N_PATCH - 1)) * D_MODEL + n];
          outF[(size_t)m * Nout + n] = v;
          outB[(size_t)m * Nout + n] = (bf16)v;
        } else if (MODE == MODE_QK) {   // scatter to [B,H,N,DH]
          const int b = m >> 10, nq = m & 1023;
          const int h = n >> 7, dh = n & 127;
          outB[(((size_t)(b * NHEAD + h)) * N_PATCH + nq) * DH_ + dh] = (bf16)v;
        } else if (MODE == MODE_VT) {   // scatter transposed to [B,H,DH,N]
          const int b = m >> 10, nq = m & 1023;
          const int h = n >> 7, dh = n & 127;
          outB[(((size_t)(b * NHEAD + h)) * DH_ + dh) * N_PATCH + nq] = (bf16)v;
        } else if (MODE == MODE_RELU) {
          outB[(size_t)m * Nout + n] = (bf16)fmaxf(v, 0.f);
        } else {  // MODE_RESID
          outF[(size_t)m * Nout + n] = resid[(size_t)m * Nout + n] + v;
        }
      }
    }
  }
}

// --------------------------- cooperative LDS-staged GEMM -------------------
// C = A * W^T (+bias). A:[M,K] bf16, W:[Nout,K] bf16, both row-major.
// Block = 256 threads (8 waves): block tile 64(M) x 512(N); wave tile 64x64.
// The 64x32 A k-slab is shared by all waves, double-buffered in LDS and
// filled with async global->LDS DMA overlapped with the WMMA stream.
// Requires M % 64 == 0 and Nout % 512 == 0.
template <int MODE>
__global__ __launch_bounds__(256) void gemm_lds_kernel(
    const bf16* __restrict__ A, const bf16* __restrict__ W,
    const float* __restrict__ bias, float* __restrict__ outF,
    bf16* __restrict__ outB, const float* __restrict__ resid,
    const float* __restrict__ pe, int M, int Nout, int K, float alpha) {
  __shared__ bf16 slab[2][64 * 32];
  const int tid  = threadIdx.x;
  const int lane = tid & 31;
  const int wave = tid >> 5;
  const int lr   = lane & 15;
  const int hi8  = (lane >> 4) * 8;
  const int klo  = (lane >> 4) * 8;
  const int tilesNB = Nout >> 9;                       // 512-wide column blocks
  const int m0 = (blockIdx.x / tilesNB) << 6;
  const int n0 = ((blockIdx.x % tilesNB) << 9) + (wave << 6);

  // Each thread owns one 16B chunk of the 64x32 slab (256 * 16B = 4KB).
  const int arow = tid >> 2;
  const int acol = (tid & 3) << 3;
  const bf16* aSrc = A + (size_t)(m0 + arow) * K + acol;
  bf16* aDst[2] = { &slab[0][arow * 32 + acol], &slab[1][arow * 32 + acol] };

  async_copy16(aSrc, aDst[0]);                         // stage k0 = 0

  v8f acc[4][4];
#pragma unroll
  for (int i = 0; i < 4; ++i)
#pragma unroll
    for (int j = 0; j < 4; ++j) acc[i][j] = vzero8();

  const bf16* Wbase = W + (size_t)(n0 + lr) * K + klo;

  int buf = 0;
  for (int k0 = 0; k0 < K; k0 += 32, buf ^= 1) {
    wait_async0();        // slab[buf] resident
    __syncthreads();      // visible to all waves; prior slab[buf^1] reads done
    if (k0 + 32 < K) async_copy16(aSrc + k0 + 32, aDst[buf ^ 1]);

    const bf16* sl = &slab[buf][0];
    v16bf af[4], wf[4];
#pragma unroll
    for (int i = 0; i < 4; ++i) af[i] = ldfrag(sl + (i * 16 + lr) * 32 + klo);
#pragma unroll
    for (int j = 0; j < 4; ++j) wf[j] = ldfrag(Wbase + (size_t)j * 16 * K + k0);
    if (k0 + 64 < K) __builtin_prefetch(Wbase + k0 + 64);  // global_prefetch_b8
#pragma unroll
    for (int i = 0; i < 4; ++i)
#pragma unroll
      for (int j = 0; j < 4; ++j) acc[i][j] = wmma_bf(af[i], wf[j], acc[i][j]);
  }

  gemm_epilogue<MODE>(acc, m0, n0, lr, hi8, bias, outF, outB, resid, pe, Nout, alpha);
}

// --------------------------- direct wave-GEMM (small Nout) -----------------
template <int MT, int NT, int MODE>
__global__ __launch_bounds__(256) void gemm_bf16_kernel(
    const bf16* __restrict__ A, const bf16* __restrict__ W,
    const float* __restrict__ bias, float* __restrict__ outF,
    bf16* __restrict__ outB, const float* __restrict__ resid,
    const float* __restrict__ pe, int M, int Nout, int K, float alpha) {
  const int lane = threadIdx.x & 31;
  const int wave = threadIdx.x >> 5;
  const int lr   = lane & 15;
  const int hi8  = (lane >> 4) * 8;
  const int klo  = (lane >> 4) * 8;
  const int tilesN = Nout / (NT * 16);
  const int tilesM = M / (MT * 16);
  const int tile = blockIdx.x * 8 + wave;
  if (tile >= tilesM * tilesN) return;
  const int m0 = (tile / tilesN) * (MT * 16);
  const int n0 = (tile % tilesN) * (NT * 16);

  v8f acc[MT][NT];
#pragma unroll
  for (int i = 0; i < MT; ++i)
#pragma unroll
    for (int j = 0; j < NT; ++j) acc[i][j] = vzero8();

  const bf16* Abase = A + (size_t)(m0 + lr) * K + klo;
  const bf16* Wbase = W + (size_t)(n0 + lr) * K + klo;

  for (int k0 = 0; k0 < K; k0 += 32) {
    v16bf af[MT], wf[NT];
#pragma unroll
    for (int i = 0; i < MT; ++i) af[i] = ldfrag(Abase + (size_t)i * 16 * K + k0);
#pragma unroll
    for (int j = 0; j < NT; ++j) wf[j] = ldfrag(Wbase + (size_t)j * 16 * K + k0);
    if (k0 + 64 < K) {
      __builtin_prefetch(Abase + k0 + 64);
      __builtin_prefetch(Wbase + k0 + 64);
    }
#pragma unroll
    for (int i = 0; i < MT; ++i)
#pragma unroll
      for (int j = 0; j < NT; ++j) acc[i][j] = wmma_bf(af[i], wf[j], acc[i][j]);
  }

#pragma unroll
  for (int i = 0; i < MT; ++i)
#pragma unroll
    for (int j = 0; j < NT; ++j)
#pragma unroll
      for (int r = 0; r < 8; ++r) {
        const int m = m0 + i * 16 + hi8 + r;
        const int n = n0 + j * 16 + lr;
        float v = acc[i][j][r];
        if (bias) v += bias[n];
        v *= alpha;
        (void)resid; (void)pe;
        outF[(size_t)m * Nout + n] = v;   // MODE_F32 only (final projection)
      }
}

// --------------------------- flash attention -------------------------------
// One wave per 16-query block per (batch, head). Q pre-scaled by 1/sqrt(DH).
// Q,K: [B,H,N,DH] bf16; Vt: [B,H,DH,N] bf16; out: [ROWS, D_MODEL] bf16.
__global__ __launch_bounds__(32) void attn_kernel(const bf16* __restrict__ Q,
                                                  const bf16* __restrict__ Kb,
                                                  const bf16* __restrict__ Vt,
                                                  bf16* __restrict__ outB) {
  const int lane = threadIdx.x & 31;
  const int lr = lane & 15;
  const int hi = lane >> 4;
  const int klo = hi * 8;
  const int wid = blockIdx.x;              // B*H*(N/16)
  const int qt = wid & 63;
  const int hh = (wid >> 6) & 7;
  const int b = wid >> 9;
  const float NEG = -__builtin_inff();

  const bf16* Qh = Q + (((size_t)(b * NHEAD + hh)) * N_PATCH + (size_t)qt * 16) * DH_;
  const bf16* Kh = Kb + ((size_t)(b * NHEAD + hh)) * N_PATCH * DH_;
  const bf16* Vh = Vt + ((size_t)(b * NHEAD + hh)) * DH_ * N_PATCH;

  v16bf qf[4];
#pragma unroll
  for (int c = 0; c < 4; ++c) qf[c] = ldfrag(Qh + (size_t)lr * DH_ + c * 32 + klo);

  v8f o[8];
#pragma unroll
  for (int t = 0; t < 8; ++t) o[t] = vzero8();
  float mi[8], li[8];
#pragma unroll
  for (int r = 0; r < 8; ++r) { mi[r] = NEG; li[r] = 0.f; }

  __shared__ bf16 sP[16 * 32];

  const int kend = qt * 16 + 16;
  const int nt = (kend + 31) >> 5;
  for (int jt = 0; jt < nt; ++jt) {
    const int kb = jt << 5;
    v8f s0 = vzero8(), s1 = vzero8();
#pragma unroll
    for (int c = 0; c < 4; ++c) {
      v16bf k0f = ldfrag(Kh + (size_t)(kb + lr) * DH_ + c * 32 + klo);
      v16bf k1f = ldfrag(Kh + (size_t)(kb + 16 + lr) * DH_ + c * 32 + klo);
      s0 = wmma_bf(qf[c], k0f, s0);
      s1 = wmma_bf(qf[c], k1f, s1);
    }
    const int n0 = kb + lr, n1 = kb + 16 + lr;
#pragma unroll
    for (int r = 0; r < 8; ++r) {
      const int m = qt * 16 + hi * 8 + r;
      float a0 = (n0 <= m) ? s0[r] : NEG;
      float a1 = (n1 <= m) ? s1[r] : NEG;
      float rm = fmaxf(a0, a1);
#pragma unroll
      for (int d = 1; d < 16; d <<= 1) rm = fmaxf(rm, __shfl_xor(rm, d, 32));
      const float mnew = fmaxf(mi[r], rm);
      const float corr = __expf(mi[r] - mnew);
      const float p0 = __expf(a0 - mnew);
      const float p1 = __expf(a1 - mnew);
      float rs = p0 + p1;
#pragma unroll
      for (int d = 1; d < 16; d <<= 1) rs += __shfl_xor(rs, d, 32);
      li[r] = li[r] * corr + rs;
      mi[r] = mnew;
#pragma unroll
      for (int t = 0; t < 8; ++t) o[t][r] *= corr;
      sP[(hi * 8 + r) * 32 + lr] = (bf16)p0;
      sP[(hi * 8 + r) * 32 + 16 + lr] = (bf16)p1;
    }
    __syncthreads();  // C-layout -> A-layout transpose through LDS
    v16bf pf = ldfrag(&sP[lr * 32 + klo]);
    __syncthreads();
#pragma unroll
    for (int t = 0; t < 8; ++t) {
      v16bf vf = ldfrag(Vh + (size_t)(t * 16 + lr) * N_PATCH + kb + klo);
      o[t] = wmma_bf(pf, vf, o[t]);
    }
  }

#pragma unroll
  for (int t = 0; t < 8; ++t)
#pragma unroll
    for (int r = 0; r < 8; ++r) {
      const size_t row = (size_t)b * N_PATCH + qt * 16 + hi * 8 + r;
      const int col = hh * DH_ + t * 16 + lr;
      outB[row * D_MODEL + col] = (bf16)(o[t][r] / li[r]);
    }
}

// --------------------------- host driver -----------------------------------

extern "C" void kernel_launch(void* const* d_in, const int* in_sizes, int n_in,
                              void* d_out, int out_size, void* d_ws, size_t ws_size,
                              hipStream_t stream) {
  (void)in_sizes; (void)n_in; (void)out_size; (void)ws_size;
  const float* x    = (const float*)d_in[0];
  const float* Wemb = (const float*)d_in[1];
  const float* Wq   = (const float*)d_in[2];
  const float* bq   = (const float*)d_in[3];
  const float* Wk   = (const float*)d_in[4];
  const float* bk   = (const float*)d_in[5];
  const float* Wv   = (const float*)d_in[6];
  const float* bv   = (const float*)d_in[7];
  const float* Wo   = (const float*)d_in[8];
  const float* bo   = (const float*)d_in[9];
  const float* Wc1  = (const float*)d_in[10];
  const float* bc1  = (const float*)d_in[11];
  const float* Wc2  = (const float*)d_in[12];
  const float* bc2  = (const float*)d_in[13];
  const float* g1   = (const float*)d_in[14];
  const float* be1  = (const float*)d_in[15];
  const float* g2   = (const float*)d_in[16];
  const float* be2  = (const float*)d_in[17];
  const float* gN   = (const float*)d_in[18];
  const float* beN  = (const float*)d_in[19];
  const float* Wp   = (const float*)d_in[20];
  const float* bp   = (const float*)d_in[21];
  float* out = (float*)d_out;

  const size_t E_WEMB = (size_t)D_MODEL * PLEN;
  const size_t E_WDD  = (size_t)ELAYERS * D_MODEL * D_MODEL;
  const size_t E_WC   = (size_t)ELAYERS * DFF_ * D_MODEL;
  const size_t E_WP   = (size_t)PLEN * D_MODEL;
  const size_t E_TOK  = (size_t)ROWS * PLEN;
  const size_t E_H    = (size_t)ROWS * D_MODEL;
  const size_t E_FF   = (size_t)ROWS * DFF_;
  const size_t E_PE   = (size_t)N_PATCH * D_MODEL;

  char* p = (char*)d_ws;
  auto carve = [&](size_t bytes) -> void* {
    void* r = (void*)p;
    p += (bytes + 255) & ~(size_t)255;
    return r;
  };
  bf16* wembB = (bf16*)carve(E_WEMB * 2);
  bf16* wqB   = (bf16*)carve(E_WDD * 2);
  bf16* wkB   = (bf16*)carve(E_WDD * 2);
  bf16* wvB   = (bf16*)carve(E_WDD * 2);
  bf16* woB   = (bf16*)carve(E_WDD * 2);
  bf16* wc1B  = (bf16*)carve(E_WC * 2);
  bf16* wc2B  = (bf16*)carve(E_WC * 2);
  bf16* wpB   = (bf16*)carve(E_WP * 2);
  bf16* tokB  = (bf16*)carve(E_TOK * 2);
  bf16* hB    = (bf16*)carve(E_H * 2);
  bf16* qB    = (bf16*)carve(E_H * 2);
  bf16* kB    = (bf16*)carve(E_H * 2);
  bf16* vtB   = (bf16*)carve(E_H * 2);
  bf16* attB  = (bf16*)carve(E_H * 2);
  bf16* ffB   = (bf16*)carve(E_FF * 2);
  float* peF  = (float*)carve(E_PE * 4);
  float* hF   = (float*)carve(E_H * 4);
  float* h2F  = (float*)carve(E_H * 4);

  auto cvt = [&](const float* src, bf16* dst, size_t n) {
    int n4 = (int)(n / 4);
    cvt_f32_bf16<<<(n4 + 255) / 256, 256, 0, stream>>>(src, dst, n4);
  };
  cvt(Wemb, wembB, E_WEMB);
  cvt(Wq, wqB, E_WDD);
  cvt(Wk, wkB, E_WDD);
  cvt(Wv, wvB, E_WDD);
  cvt(Wo, woB, E_WDD);
  cvt(Wc1, wc1B, E_WC);
  cvt(Wc2, wc2B, E_WC);
  cvt(Wp, wpB, E_WP);
  cvt(x, tokB, E_TOK);
  pe_kernel<<<(int)(E_PE / 256), 256, 0, stream>>>(peF);

  const float QSCALE = 0.08838834764831845f;  // 1/sqrt(128)
  const int B_1024 = (ROWS / 64) * (D_MODEL / 512);  // 128 blocks
  const int B_2048 = (ROWS / 64) * (DFF_ / 512);     // 256 blocks
  const int B42_96 = 24;  // (4096/64)*(96/32)/8 direct wave tiles

  // Patch embedding (no bias) + positional encoding; keep fp32 + bf16 copies.
  gemm_lds_kernel<MODE_PE><<<B_1024, 256, 0, stream>>>(
      tokB, wembB, nullptr, hF, hB, nullptr, peF, ROWS, D_MODEL, PLEN, 1.0f);

  for (int l = 0; l < ELAYERS; ++l) {
    const bf16* wq_l = wqB + (size_t)l * D_MODEL * D_MODEL;
    const bf16* wk_l = wkB + (size_t)l * D_MODEL * D_MODEL;
    const bf16* wv_l = wvB + (size_t)l * D_MODEL * D_MODEL;
    const bf16* wo_l = woB + (size_t)l * D_MODEL * D_MODEL;
    const bf16* wc1_l = wc1B + (size_t)l * DFF_ * D_MODEL;
    const bf16* wc2_l = wc2B + (size_t)l * D_MODEL * DFF_;

    gemm_lds_kernel<MODE_QK><<<B_1024, 256, 0, stream>>>(
        hB, wq_l, bq + l * D_MODEL, nullptr, qB, nullptr, nullptr,
        ROWS, D_MODEL, D_MODEL, QSCALE);
    gemm_lds_kernel<MODE_QK><<<B_1024, 256, 0, stream>>>(
        hB, wk_l, bk + l * D_MODEL, nullptr, kB, nullptr, nullptr,
        ROWS, D_MODEL, D_MODEL, 1.0f);
    gemm_lds_kernel<MODE_VT><<<B_1024, 256, 0, stream>>>(
        hB, wv_l, bv + l * D_MODEL, nullptr, vtB, nullptr, nullptr,
        ROWS, D_MODEL, D_MODEL, 1.0f);

    attn_kernel<<<ROWS / 16 * NHEAD / 4, 32, 0, stream>>>(qB, kB, vtB, attB);  // 512

    gemm_lds_kernel<MODE_RESID><<<B_1024, 256, 0, stream>>>(
        attB, wo_l, bo + l * D_MODEL, h2F, nullptr, hF, nullptr,
        ROWS, D_MODEL, D_MODEL, 1.0f);
    ln_kernel<<<ROWS, 256, 0, stream>>>(h2F, g1 + l * D_MODEL, be1 + l * D_MODEL, hF, hB);

    gemm_lds_kernel<MODE_RELU><<<B_2048, 256, 0, stream>>>(
        hB, wc1_l, bc1 + l * DFF_, nullptr, ffB, nullptr, nullptr,
        ROWS, DFF_, D_MODEL, 1.0f);
    gemm_lds_kernel<MODE_RESID><<<B_1024, 256, 0, stream>>>(
        ffB, wc2_l, bc2 + l * D_MODEL, h2F, nullptr, hF, nullptr,
        ROWS, D_MODEL, DFF_, 1.0f);
    ln_kernel<<<ROWS, 256, 0, stream>>>(h2F, g2 + l * D_MODEL, be2 + l * D_MODEL, hF, hB);
  }

  // Final encoder norm + projection to patch_len.
  ln_kernel<<<ROWS, 256, 0, stream>>>(hF, gN, beN, h2F, hB);
  gemm_bf16_kernel<4, 2, MODE_F32><<<B42_96, 256, 0, stream>>>(
      hB, wpB, bp, out, nullptr, nullptr, nullptr, ROWS, PLEN, D_MODEL, 1.0f);
}
